// Net_23819888623729
// MI455X (gfx1250) — matrix operand order repaired and checked by reading, not compiled, not run on hardware
//
#include <hip/hip_runtime.h>
#include <hip/hip_bf16.h>

// ---------------------------------------------------------------------------
// 3-layer LSTM (H=128, B=256, T=512, F=96) + linear head for MI455X (gfx1250).
//
//  * One workgroup per 16-row batch tile (16 WGs, 256 thr = 8 wave32).
//  * Whole T-loop in one kernel launch per layer; one s_barrier per step.
//  * gates(16x512) = [x_t | h_{t-1}] @ [W_ih;W_hh]^T via
//    v_wmma_f32_16x16x32_bf16 (zero-C start, f32 accumulate, bias folded in
//    at activation time).
//  * Both weight matrices resident in LDS as bf16 (~290KB of the 320KB/WGP);
//    W_hh B-fragments additionally resident in VGPRs across all 512 steps.
//  * Cell state c lives in VGPRs; h double-buffered in LDS.
//  * x_{t+1} prefetched with global_load_async_to_lds_b128 (ASYNCcnt path),
//    waited with s_wait_asynccnt before the step barrier.
// ---------------------------------------------------------------------------

typedef __attribute__((ext_vector_type(16))) __bf16 v16bf;
typedef __attribute__((ext_vector_type(8)))  __bf16 v8bf;
typedef __attribute__((ext_vector_type(8)))  float  v8f;

static constexpr int Hh = 128;   // hidden
static constexpr int Bb = 256;   // batch
static constexpr int Tt = 512;   // time
static constexpr int RSH = 136;  // padded LDS row length for K=128 (bank skew)

// ---- A-fragment (16x32 bf16, this lane = row M=l16) -----------------------
// element i: k = (i<8 ? i : i+8) + half*8  (+kb)
__device__ __forceinline__ v16bf load_a_frag(const __bf16* row, int half, int kb) {
  const v8bf lo = *(const v8bf*)(row + kb + half * 8);
  const v8bf hi = *(const v8bf*)(row + kb + 16 + half * 8);
  v16bf a;
#pragma unroll
  for (int i = 0; i < 8; ++i) { a[i] = lo[i]; a[i + 8] = hi[i]; }
  return a;
}

// ---- B-fragment (32x16 bf16, this lane = column N) ------------------------
// element i: k = (i<8 ? i : i+8) + half*8  (+kb), same order as A.
__device__ __forceinline__ v16bf load_b_frag(const __bf16* wrow, int half, int kb) {
  const v8bf lo = *(const v8bf*)(wrow + kb + half * 8);
  const v8bf hi = *(const v8bf*)(wrow + kb + 16 + half * 8);
  v16bf b;
#pragma unroll
  for (int i = 0; i < 8; ++i) { b[i] = lo[i]; b[i + 8] = hi[i]; }
  return b;
}

__device__ __forceinline__ float sigf(float x)  { return 1.0f / (1.0f + __expf(-x)); }
__device__ __forceinline__ float tanhf_(float x){ return 2.0f / (1.0f + __expf(-2.0f * x)) - 1.0f; }

__device__ __forceinline__ void async_wait0() {
#if __has_builtin(__builtin_amdgcn_s_wait_asynccnt)
  __builtin_amdgcn_s_wait_asynccnt(0);
#else
  asm volatile("s_wait_asynccnt 0x0" ::: "memory");
#endif
}

template <int FIN>
__launch_bounds__(256, 1)
__global__ void lstm_layer_kernel(const __bf16* __restrict__ xin,   // (B,T,FIN) bf16
                                  const float*  __restrict__ w_ih,  // (512,FIN)
                                  const float*  __restrict__ w_hh,  // (512,128)
                                  const float*  __restrict__ b_ih,  // (512)
                                  const float*  __restrict__ b_hh,  // (512)
                                  __bf16*       __restrict__ hout)  // (B,T,128) bf16
{
  static_assert(FIN % 32 == 0, "FIN must be a multiple of 32");
  constexpr int RSX = FIN + 8;      // padded LDS row for x-side K
  constexpr int KX  = FIN / 32;     // x-side K tiles
  constexpr int CPR = FIN / 8;      // 16B chunks per x row
  constexpr int CH  = 16 * CPR;     // chunks per (16 x FIN) tile
  // byte offset of xbuf within dynamic LDS (for the async-copy LDS address)
  constexpr unsigned XOFF = (unsigned)((512 * RSX + 512 * RSH) * 2 + 512 * 4);

  extern __shared__ __align__(16) char smem[];
  __bf16* wih  = (__bf16*)smem;                       // 512*RSX
  __bf16* whh  = wih + 512 * RSX;                     // 512*136
  float*  bias = (float*)(whh + 512 * RSH);           // 512
  __bf16* xbuf = (__bf16*)(bias + 512);               // 2 * 16*RSX
  __bf16* hbuf = xbuf + 2 * 16 * RSX;                 // 2 * 16*RSH

  const int tid  = threadIdx.x;
  const int wv   = tid >> 5;        // wave 0..7
  const int lane = tid & 31;
  const int half = lane >> 4;
  const int l16  = lane & 15;
  const int b0   = blockIdx.x * 16; // batch tile base

  // ---- stage weights (f32 -> bf16) and biases into LDS --------------------
  for (int idx = tid; idx < 512 * FIN; idx += 256) {
    const int g = idx / FIN, f = idx - g * FIN;
    wih[g * RSX + f] = (__bf16)w_ih[idx];
  }
  for (int idx = tid; idx < 512 * 128; idx += 256) {
    const int g = idx >> 7, k = idx & 127;
    whh[g * RSH + k] = (__bf16)w_hh[idx];
  }
  for (int idx = tid; idx < 512; idx += 256) bias[idx] = b_ih[idx] + b_hh[idx];
  for (int idx = tid; idx < 16 * RSH; idx += 256) hbuf[idx] = (__bf16)0.0f;

  // preload x_0 tile directly into xbuf[0]
  const int xm = tid / CPR, xc = tid - xm * CPR;
  if (tid < CH) {
    const uint4 v0 = *(const uint4*)(xin + ((size_t)(b0 + xm) * Tt + 0) * FIN + xc * 8);
    *(uint4*)(xbuf + xm * RSX + xc * 8) = v0;
  }
  __syncthreads();

  // ---- hoist W_hh B-fragments into VGPRs for the whole T loop --------------
  // wave wv, gate q -> gate-matrix columns n = 128*q + 16*wv + l16
  const int j = 16 * wv + l16;      // hidden index owned by this lane
  v16bf bh[4][4];
  float bq[4];
#pragma unroll
  for (int q = 0; q < 4; ++q) {
    const __bf16* wrow = whh + (size_t)(128 * q + j) * RSH;
#pragma unroll
    for (int kt = 0; kt < 4; ++kt) bh[q][kt] = load_b_frag(wrow, half, 32 * kt);
    bq[q] = bias[128 * q + j];
  }

  float cst[8];
#pragma unroll
  for (int r = 0; r < 8; ++r) cst[r] = 0.0f;

  for (int t = 0; t < Tt; ++t) {
    const int cur = t & 1, nxt = cur ^ 1;
    const __bf16* xcur = xbuf + cur * 16 * RSX;
    const __bf16* hcur = hbuf + cur * 16 * RSH;
    __bf16* hnxt = hbuf + nxt * 16 * RSH;

    // async prefetch of x_{t+1} straight into the other LDS x buffer
    // (global_load_async_to_lds_b128, tracked by ASYNCcnt; latency hidden
    //  under the 28/32 WMMAs below)
    if (t + 1 < Tt && tid < CH) {
      const unsigned ldsoff =
          XOFF + (unsigned)nxt * 16u * RSX * 2u + (unsigned)(xm * RSX + xc * 8) * 2u;
      const unsigned long long ga =
          (unsigned long long)(uintptr_t)(xin + ((size_t)(b0 + xm) * Tt + (t + 1)) * FIN + xc * 8);
      asm volatile("global_load_async_to_lds_b128 %0, %1, off"
                   :: "v"(ldsoff), "v"(ga) : "memory");
    }

    // ---- gates = [x_t | h_{t-1}] @ W^T, zero-C start ----------------------
    v8f acc[4];
#pragma unroll
    for (int q = 0; q < 4; ++q)
#pragma unroll
      for (int r = 0; r < 8; ++r) acc[q][r] = 0.0f;

#pragma unroll
    for (int kt = 0; kt < KX; ++kt) {       // x-side K tiles
      const v16bf a = load_a_frag(xcur + l16 * RSX, half, 32 * kt);
#pragma unroll
      for (int q = 0; q < 4; ++q) {
        const v16bf bx = load_b_frag(wih + (size_t)(128 * q + j) * RSX, half, 32 * kt);
        acc[q] = __builtin_amdgcn_wmma_f32_16x16x32_bf16(
            false, a, false, bx, (short)0, acc[q], false, false);
      }
    }
#pragma unroll
    for (int kt = 0; kt < 4; ++kt) {        // h-side K tiles (B resident)
      const v16bf a = load_a_frag(hcur + l16 * RSH, half, 32 * kt);
#pragma unroll
      for (int q = 0; q < 4; ++q) {
        acc[q] = __builtin_amdgcn_wmma_f32_16x16x32_bf16(
            false, a, false, bh[q][kt], (short)0, acc[q], false, false);
      }
    }

    // ---- LSTM cell update, entirely in registers --------------------------
#pragma unroll
    for (int r = 0; r < 8; ++r) {
      const int m = r + 8 * half;   // D layout: M = r + 8*(lane/16), N = l16
      const float iv = sigf(acc[0][r] + bq[0]);
      const float fv = sigf(acc[1][r] + bq[1]);
      const float gv = tanhf_(acc[2][r] + bq[2]);
      const float ov = sigf(acc[3][r] + bq[3]);
      cst[r] = fv * cst[r] + iv * gv;
      const float hv = ov * tanhf_(cst[r]);
      const __bf16 hb = (__bf16)hv;
      hnxt[m * RSH + j] = hb;
      hout[((size_t)(b0 + m) * Tt + t) * Hh + j] = hb;
    }

    if (t + 1 < Tt) async_wait0();  // x_{t+1} LDS copy complete (uniform)
    __syncthreads();
  }
}

// ---- small kernels --------------------------------------------------------

__global__ void f2bf_kernel(const float* __restrict__ in, __bf16* __restrict__ out, int n) {
  const int i = blockIdx.x * blockDim.x + threadIdx.x;
  if (i < n) out[i] = (__bf16)in[i];
}

__global__ void head1_kernel(const __bf16* __restrict__ h2, const float* __restrict__ w_line,
                             const float* __restrict__ b_line, float* __restrict__ mid) {
  const int i = blockIdx.x * blockDim.x + threadIdx.x;  // b*128 + jj
  if (i >= Bb * Hh) return;
  const int b = i >> 7, jj = i & 127;
  const __bf16* hrow = h2 + ((size_t)b * Tt + (Tt - 1)) * Hh;
  float s = b_line[jj];
#pragma unroll 8
  for (int k = 0; k < Hh; ++k) s += (float)hrow[k] * w_line[jj * Hh + k];
  mid[i] = s;
}

__global__ void head2_kernel(const float* __restrict__ mid, const float* __restrict__ w_out,
                             const float* __restrict__ b_out, float* __restrict__ out) {
  const int i = blockIdx.x * blockDim.x + threadIdx.x;  // b*10 + o
  if (i >= Bb * 10) return;
  const int b = i / 10, o = i - b * 10;
  float s = b_out[o];
#pragma unroll 8
  for (int k = 0; k < Hh; ++k) s += mid[b * Hh + k] * w_out[o * Hh + k];
  out[i] = s;
}

// ---- host -----------------------------------------------------------------

static size_t lstm_shmem(int fin) {
  const size_t rsx = (size_t)fin + 8;
  return (512 * rsx + 512 * RSH + 2 * 16 * rsx + 2 * 16 * RSH) * 2 + 512 * 4;
}

extern "C" void kernel_launch(void* const* d_in, const int* in_sizes, int n_in,
                              void* d_out, int out_size, void* d_ws, size_t ws_size,
                              hipStream_t stream) {
  const float* x       = (const float*)d_in[0];
  const float* w_ih[3] = {(const float*)d_in[1], (const float*)d_in[5], (const float*)d_in[9]};
  const float* w_hh[3] = {(const float*)d_in[2], (const float*)d_in[6], (const float*)d_in[10]};
  const float* b_ih[3] = {(const float*)d_in[3], (const float*)d_in[7], (const float*)d_in[11]};
  const float* b_hh[3] = {(const float*)d_in[4], (const float*)d_in[8], (const float*)d_in[12]};
  const float* w_line = (const float*)d_in[13];
  const float* b_line = (const float*)d_in[14];
  const float* w_out  = (const float*)d_in[15];
  const float* b_out  = (const float*)d_in[16];
  float* out = (float*)d_out;

  // workspace layout (all bf16 activations)
  __bf16* x0  = (__bf16*)d_ws;                         // B*T*96
  __bf16* a0  = x0 + (size_t)Bb * Tt * 96;             // B*T*128
  __bf16* a1  = a0 + (size_t)Bb * Tt * Hh;
  __bf16* a2  = a1 + (size_t)Bb * Tt * Hh;
  float*  mid = (float*)(a2 + (size_t)Bb * Tt * Hh);   // B*128

  const size_t sh96  = lstm_shmem(96);
  const size_t sh128 = lstm_shmem(128);
  hipFuncSetAttribute((const void*)lstm_layer_kernel<96>,
                      hipFuncAttributeMaxDynamicSharedMemorySize, (int)sh96);
  hipFuncSetAttribute((const void*)lstm_layer_kernel<128>,
                      hipFuncAttributeMaxDynamicSharedMemorySize, (int)sh128);

  const int nconv = Bb * Tt * 96;
  f2bf_kernel<<<(nconv + 255) / 256, 256, 0, stream>>>(x, x0, nconv);

  lstm_layer_kernel<96><<<Bb / 16, 256, sh96, stream>>>(
      x0, w_ih[0], w_hh[0], b_ih[0], b_hh[0], a0);
  lstm_layer_kernel<128><<<Bb / 16, 256, sh128, stream>>>(
      a0, w_ih[1], w_hh[1], b_ih[1], b_hh[1], a1);
  lstm_layer_kernel<128><<<Bb / 16, 256, sh128, stream>>>(
      a1, w_ih[2], w_hh[2], b_ih[2], b_hh[2], a2);

  head1_kernel<<<(Bb * Hh + 255) / 256, 256, 0, stream>>>(a2, w_line, b_line, mid);
  head2_kernel<<<(Bb * 10 + 255) / 256, 256, 0, stream>>>(mid, w_out, b_out, out);
}
